// RoPEMultiHeadSelfAttention_90649579749828
// MI455X (gfx1250) — matrix operand order repaired and checked by reading, not compiled, not run on hardware
//
#include <hip/hip_runtime.h>

typedef __attribute__((ext_vector_type(16))) _Float16 v16h;
typedef __attribute__((ext_vector_type(8)))  _Float16 v8h;
typedef __attribute__((ext_vector_type(8)))  float    v8f;

#if defined(__has_builtin)
#if __has_builtin(__builtin_amdgcn_tensor_load_to_lds) && \
    __has_builtin(__builtin_amdgcn_s_wait_tensorcnt)
#define HAVE_TDM 1
#endif
#endif

#ifdef HAVE_TDM
typedef __attribute__((ext_vector_type(4))) unsigned int u32x4;
typedef __attribute__((ext_vector_type(8))) int          i32x8;
typedef __attribute__((ext_vector_type(4))) int          i32x4;
#endif

// ---------------------------------------------------------------------------
// WMMA fragment loaders (layouts per CDNA5 ISA 7.12.2, f16, wave32)
//  A (16x32, MxK): lane L holds row M=L%16; lanes<16: K {0..7}U{16..23},
//                  lanes>=16: K {8..15}U{24..31}.
//  B (32x16, KxN): lane L holds col N=L%16; lanes<16: K {0..7}U{16..23},
//                  lanes>=16: K {8..15}U{24..31}.
//  C/D (16x16):    VGPR r: lanes 0-15 -> M=r, lanes 16-31 -> M=r+8; N=lane%16.
// ---------------------------------------------------------------------------
__device__ inline v16h frag_a(const _Float16* base, int ld, int koff, int lane) {
  const _Float16* p = base + (lane & 15) * ld + koff + ((lane >> 4) << 3);
  v16h r;
  ((v8h*)&r)[0] = *(const v8h*)(p);
  ((v8h*)&r)[1] = *(const v8h*)(p + 16);
  return r;
}

__device__ inline v16h frag_b(const _Float16* base, int ld, int koff, int lane) {
  const _Float16* p = base + (lane & 15) * ld + koff + ((lane >> 4) << 3);
  v16h r;
  ((v8h*)&r)[0] = *(const v8h*)(p);
  ((v8h*)&r)[1] = *(const v8h*)(p + 16);
  return r;
}

// ---------------------------------------------------------------------------
// Double-buffered f32->f16 WMMA GEMM: C[M,N] = A[M,K] * W[N,K]^T (row-major)
// 256 thr (8 waves), tile 128x128x32, wave grid 4x2 (32x64 per wave).
// EPI==0: f32 C to out[m*N+n].  EPI==1: QKV scatter -> f16 (which,b,h,s,d).
// ---------------------------------------------------------------------------
__device__ inline void g_load16(const float* __restrict__ src, bool valid,
                                float4 f[4]) {
  if (valid) {
    const float4* p = reinterpret_cast<const float4*>(src);
    f[0] = p[0]; f[1] = p[1]; f[2] = p[2]; f[3] = p[3];
  } else {
    f[0] = f[1] = f[2] = f[3] = make_float4(0.f, 0.f, 0.f, 0.f);
  }
}

__device__ inline void cvt_store16(_Float16* dst, const float4 f[4]) {
  const float* fp = reinterpret_cast<const float*>(f);
  v8h h0, h1;
#pragma unroll
  for (int i = 0; i < 8; i++) { h0[i] = (_Float16)fp[i]; h1[i] = (_Float16)fp[8 + i]; }
  *(v8h*)dst = h0;
  *(v8h*)(dst + 8) = h1;
}

template<int EPI>
__global__ void __launch_bounds__(256)
gemm_wmma_kernel(const float* __restrict__ A, const float* __restrict__ W,
                 void* __restrict__ outp, int M, int N, int K) {
  constexpr int BM = 128, BN = 128, BK = 32, LDT = 40;  // padded LDS rows
  __shared__ _Float16 As[2][BM * LDT];
  __shared__ _Float16 Bs[2][BN * LDT];

  const int tid  = threadIdx.x;
  const int lane = tid & 31;
  const int wave = tid >> 5;
  const int wrow = (wave & 3) * 32;
  const int wcol = (wave >> 2) * 64;
  const int n0 = blockIdx.x * BN;
  const int m0 = blockIdx.y * BM;

  const v8f vzero = {0.f, 0.f, 0.f, 0.f, 0.f, 0.f, 0.f, 0.f};
  v8f acc[2][4];
#pragma unroll
  for (int i = 0; i < 2; i++)
#pragma unroll
    for (int j = 0; j < 4; j++) acc[i][j] = vzero;

  const int row = tid >> 1;        // 0..127
  const int seg = (tid & 1) << 4;  // 0 or 16 floats within the 32-col row
  const int gm = m0 + row;
  const int gn = n0 + row;
  const bool va = gm < M, vb = gn < N;
  const float* ap = A + (size_t)gm * K + seg;
  const float* wp = W + (size_t)gn * K + seg;

  float4 fa[4], fb[4];
  // prologue: stage k-tile 0 into buffer 0
  g_load16(ap, va, fa);
  g_load16(wp, vb, fb);
  cvt_store16(&As[0][row * LDT + seg], fa);
  cvt_store16(&Bs[0][row * LDT + seg], fb);
  __syncthreads();

  const int nk = K / BK;
  for (int kt = 0; kt < nk; kt++) {
    const int cur = kt & 1, nxt = cur ^ 1;
    const bool more = (kt + 1) < nk;
    if (more) {  // issue next tile's global loads before the WMMA chain
      g_load16(ap + (kt + 1) * BK, va, fa);
      g_load16(wp + (kt + 1) * BK, vb, fb);
      __builtin_prefetch(ap + (kt + 2) * BK, 0, 1);
      __builtin_prefetch(wp + (kt + 2) * BK, 0, 1);
    }

    v16h af[2], bf[4];
#pragma unroll
    for (int i = 0; i < 2; i++) af[i] = frag_a(&As[cur][(wrow + i * 16) * LDT], LDT, 0, lane);
#pragma unroll
    for (int j = 0; j < 4; j++) bf[j] = frag_b(&Bs[cur][(wcol + j * 16) * LDT], LDT, 0, lane);
#pragma unroll
    for (int i = 0; i < 2; i++)
#pragma unroll
      for (int j = 0; j < 4; j++)
        acc[i][j] = __builtin_amdgcn_wmma_f32_16x16x32_f16(
            false, af[i], false, bf[j], (short)0, acc[i][j], false, false);

    if (more) {
      cvt_store16(&As[nxt][row * LDT + seg], fa);
      cvt_store16(&Bs[nxt][row * LDT + seg], fb);
      __syncthreads();
    }
  }

  const int hi8 = (lane >> 4) << 3;
  if (EPI == 0) {
    float* out = (float*)outp;
#pragma unroll
    for (int i = 0; i < 2; i++)
#pragma unroll
      for (int j = 0; j < 4; j++)
#pragma unroll
        for (int r = 0; r < 8; r++) {
          const int m = m0 + wrow + i * 16 + r + hi8;
          const int n = n0 + wcol + j * 16 + (lane & 15);
          if (m < M) out[(size_t)m * N + n] = acc[i][j][r];
        }
  } else {
    _Float16* qkv = (_Float16*)outp;
#pragma unroll
    for (int i = 0; i < 2; i++)
#pragma unroll
      for (int j = 0; j < 4; j++)
#pragma unroll
        for (int r = 0; r < 8; r++) {
          const int m = m0 + wrow + i * 16 + r + hi8;
          const int n = n0 + wcol + j * 16 + (lane & 15);
          if (m < M) {
            const int b = m / 577, s = m - b * 577;
            const int which = n / 768, rem = n - which * 768;
            const int h = rem >> 6, d = rem & 63;
            const size_t idx =
                ((size_t)which * (32 * 12) + (size_t)b * 12 + h) * (577 * 64) +
                (size_t)s * 64 + d;
            qkv[idx] = (_Float16)acc[i][j][r];
          }
        }
  }
}

// ---------------------------------------------------------------------------
// RoPE: f16 planes (which,b,h,s,d); which==2 (V) and s==0 (CLS) copy through.
// ---------------------------------------------------------------------------
__global__ void __launch_bounds__(256)
rope_kernel(const _Float16* __restrict__ in, _Float16* __restrict__ out) {
  constexpr unsigned PL = 32u * 12u * 577u * 64u;
  const unsigned idx = blockIdx.x * 256u + threadIdx.x;
  if (idx >= 3u * PL) return;
  const unsigned which = idx / PL;
  const unsigned wi = idx - which * PL;
  const unsigned d = wi & 63u;
  const unsigned s = (wi >> 6) % 577u;
  if (which == 2u || s == 0u) { out[idx] = in[idx]; return; }

  const int p  = (int)s - 1;
  const int iy = p / 24;
  const int jx = p - iy * 24;
  const int half = (int)(d >> 5);
  const int dd   = (int)(d & 31u);
  const float nc = (half == 0 ? (float)jx : (float)iy) * (1.0f / 23.0f);
  const float f2 = (float)(2 * (dd >> 1));
  const float divt = __expf(-f2 * 0.28782313662425572f);  // ln(1e4)/32
  const float ang = nc * divt;
  const float c = __cosf(ang), sn = __sinf(ang);
  const unsigned base = idx - (unsigned)dd;
  const float t0 = (float)in[idx];
  const float t1 = (float)in[base + (unsigned)((dd - 1) & 31)];
  out[idx] = (_Float16)(t0 * c + t1 * sn);
}

// ---------------------------------------------------------------------------
// Flash attention per (b,h,query-tile of 64). 128 thr (4 waves), 16 q-rows
// per wave. K tile staged by the Tensor Data Mover (zero-fill past S via
// tensor_dim1, LDS row padding via pad_enable); V staged transposed manually.
// ---------------------------------------------------------------------------
__global__ void __launch_bounds__(128)
attn_kernel(const _Float16* __restrict__ q, const _Float16* __restrict__ k,
            const _Float16* __restrict__ v, float* __restrict__ out) {
  constexpr int S = 577, H = 12, Dh = 64, LDK = 72;
  __shared__ _Float16 Ks[64 * LDK];     // K tile  [key][d]   (padded rows)
  __shared__ _Float16 Vt[64 * LDK];     // V tile transposed [d][key]
  __shared__ _Float16 Ps[4][16 * LDK];  // per-wave P staging [row][key]

  const int tid  = threadIdx.x;
  const int lane = tid & 31;
  const int wave = tid >> 5;
  const int qt = blockIdx.x, h = blockIdx.y, b = blockIdx.z;
  const int q0 = qt * 64;
  const size_t plane = ((size_t)b * H + h) * (size_t)S * Dh;

  // Q fragments: 16x64 per wave -> two 16x32 A-fragments, register-resident
  v16h qf[2];
  {
    const int s  = q0 + wave * 16 + (lane & 15);
    const int hi = lane >> 4;
    const v8h z8 = {0, 0, 0, 0, 0, 0, 0, 0};
#pragma unroll
    for (int kt = 0; kt < 2; kt++) {
      if (s < S) {
        const _Float16* p = q + plane + (size_t)s * Dh + kt * 32 + hi * 8;
        ((v8h*)&qf[kt])[0] = *(const v8h*)(p);
        ((v8h*)&qf[kt])[1] = *(const v8h*)(p + 16);
      } else {
        ((v8h*)&qf[kt])[0] = z8;
        ((v8h*)&qf[kt])[1] = z8;
      }
    }
  }

  const v8f vzero = {0.f, 0.f, 0.f, 0.f, 0.f, 0.f, 0.f, 0.f};
  v8f o[4];
#pragma unroll
  for (int j = 0; j < 4; j++) o[j] = vzero;
  float mi[8], li[8];
#pragma unroll
  for (int r = 0; r < 8; r++) { mi[r] = -1.0e30f; li[r] = 0.f; }

  const int lrow = tid >> 1;        // 0..63
  const int segc = (tid & 1) << 5;  // 0/32 halves

  for (int st = 0; st < 10; st++) {
    const int s0 = st * 64;
    __syncthreads();  // previous tile fully consumed before restaging

    // ---- stage K tile ----
#ifdef HAVE_TDM
    if (wave == 0) {
      const unsigned lds_off = (unsigned)(unsigned long long)(size_t)&Ks[0];
      const unsigned long long ga =
          (unsigned long long)(size_t)(k + plane + (size_t)s0 * Dh);
      const int rows = (S - s0) < 64 ? (S - s0) : 64;
      u32x4 g0;
      g0[0] = 1u;                                   // count=1, user mode
      g0[1] = lds_off;                              // lds_addr (bytes)
      g0[2] = (unsigned)(ga & 0xffffffffu);         // global_addr[31:0]
      g0[3] = (unsigned)((ga >> 32) & 0x1ffffffu)   // global_addr[56:32]
              | (2u << 30);                         // type = image
      i32x8 g1;
      g1[0] = (int)((1u << 16)      // data_size = 2 bytes
                    | (1u << 20)    // pad_enable
                    | (4u << 22)    // pad_interval: 32 DWORDs (=128B row)
                    | (3u << 25));  // pad_amount: 4 DWORDs (=16B)
      g1[1] = (int)(64u << 16);     // tensor_dim0 = 64 (low half)
      g1[2] = (int)(((unsigned)rows) << 16);  // tensor_dim1 = valid rows
      g1[3] = (int)(64u << 16);     // tile_dim0 = 64
      g1[4] = 64;                   // tile_dim1 = 64 (OOB rows -> zero)
      g1[5] = 64;                   // tensor_dim0_stride = 64 elements
      g1[6] = 0;
      g1[7] = 0;
      const i32x4 gz4 = {0, 0, 0, 0};
      const i32x8 gz8 = {0, 0, 0, 0, 0, 0, 0, 0};
      // 6-arg form (amdgpu-toolchain / therock headers): groups 0..3 + extra + cpol
      __builtin_amdgcn_tensor_load_to_lds(g0, g1, gz4, gz4, gz8, 0);
      __builtin_amdgcn_s_wait_tensorcnt((short)0);
    }
#else
    {
      const int s = s0 + lrow;
      v8h f0, f1, f2, f3;
      const v8h z8 = {0, 0, 0, 0, 0, 0, 0, 0};
      if (s < S) {
        const v8h* p = (const v8h*)(k + plane + (size_t)s * Dh + segc);
        f0 = p[0]; f1 = p[1]; f2 = p[2]; f3 = p[3];
      } else { f0 = f1 = f2 = f3 = z8; }
      _Float16* dst = &Ks[lrow * LDK + segc];
      ((v8h*)dst)[0] = f0; ((v8h*)dst)[1] = f1;
      ((v8h*)dst)[2] = f2; ((v8h*)dst)[3] = f3;
    }
#endif

    // ---- stage V tile, transposed ----
    {
      const int s = s0 + lrow;
      _Float16 tmp[32];
      if (s < S) {
        const v8h* p = (const v8h*)(v + plane + (size_t)s * Dh + segc);
        ((v8h*)tmp)[0] = p[0]; ((v8h*)tmp)[1] = p[1];
        ((v8h*)tmp)[2] = p[2]; ((v8h*)tmp)[3] = p[3];
      } else {
#pragma unroll
        for (int c = 0; c < 32; c++) tmp[c] = (_Float16)0.f;
      }
#pragma unroll
      for (int c = 0; c < 32; c++) Vt[(segc + c) * LDK + lrow] = tmp[c];
    }
    __syncthreads();

    // ---- scores = Q K^T ----
    v8f sc[4];
#pragma unroll
    for (int j = 0; j < 4; j++) sc[j] = vzero;
#pragma unroll
    for (int kt = 0; kt < 2; kt++) {
      v16h kb[4];
#pragma unroll
      for (int j = 0; j < 4; j++)
        kb[j] = frag_b(&Ks[(j * 16) * LDK], LDK, kt * 32, lane);
#pragma unroll
      for (int j = 0; j < 4; j++)
        sc[j] = __builtin_amdgcn_wmma_f32_16x16x32_f16(
            false, qf[kt], false, kb[j], (short)0, sc[j], false, false);
    }

    // ---- scale + key mask ----
#pragma unroll
    for (int j = 0; j < 4; j++) {
      const int key = s0 + j * 16 + (lane & 15);
      const bool bad = key >= S;
#pragma unroll
      for (int r = 0; r < 8; r++) {
        float x = sc[j][r] * 0.125f;  // Dh^-0.5
        sc[j][r] = bad ? -3.0e38f : x;
      }
    }

    // ---- online softmax (rows live in 16-lane groups) ----
#pragma unroll
    for (int r = 0; r < 8; r++) {
      float tm = fmaxf(fmaxf(sc[0][r], sc[1][r]), fmaxf(sc[2][r], sc[3][r]));
      tm = fmaxf(tm, __shfl_xor(tm, 1, 32));
      tm = fmaxf(tm, __shfl_xor(tm, 2, 32));
      tm = fmaxf(tm, __shfl_xor(tm, 4, 32));
      tm = fmaxf(tm, __shfl_xor(tm, 8, 32));
      const float mn = fmaxf(mi[r], tm);
      const float corr = __expf(mi[r] - mn);
      float rs = 0.f;
#pragma unroll
      for (int j = 0; j < 4; j++) {
        const float pr = __expf(sc[j][r] - mn);
        sc[j][r] = pr;
        rs += pr;
      }
      rs += __shfl_xor(rs, 1, 32);
      rs += __shfl_xor(rs, 2, 32);
      rs += __shfl_xor(rs, 4, 32);
      rs += __shfl_xor(rs, 8, 32);
      li[r] = li[r] * corr + rs;
      mi[r] = mn;
#pragma unroll
      for (int j = 0; j < 4; j++) o[j][r] = o[j][r] * corr;
    }

    // ---- stage P (wave-private; wave LDS ops are in-order) ----
    {
      const int hi8 = (lane >> 4) << 3;
#pragma unroll
      for (int j = 0; j < 4; j++)
#pragma unroll
        for (int r = 0; r < 8; r++)
          Ps[wave][(r + hi8) * LDK + j * 16 + (lane & 15)] = (_Float16)sc[j][r];
    }

    // ---- O += P * V ----
#pragma unroll
    for (int kt = 0; kt < 2; kt++) {
      const v16h pf = frag_a(&Ps[wave][0], LDK, kt * 32, lane);
      v16h vb[4];
#pragma unroll
      for (int j = 0; j < 4; j++)
        vb[j] = frag_b(&Vt[(j * 16) * LDK], LDK, kt * 32, lane);
#pragma unroll
      for (int j = 0; j < 4; j++)
        o[j] = __builtin_amdgcn_wmma_f32_16x16x32_f16(
            false, pf, false, vb[j], (short)0, o[j], false, false);
    }
  }

  // ---- epilogue: O / l -> (B,S,E) f32 ----
  const int hi8 = (lane >> 4) << 3;
#pragma unroll
  for (int j = 0; j < 4; j++)
#pragma unroll
    for (int r = 0; r < 8; r++) {
      const int s = q0 + wave * 16 + r + hi8;
      if (s < S) {
        const float inv = 1.0f / li[r];
        out[((size_t)b * S + s) * 768 + h * 64 + j * 16 + (lane & 15)] =
            o[j][r] * inv;
      }
    }
}

// ---------------------------------------------------------------------------
extern "C" void kernel_launch(void* const* d_in, const int* in_sizes, int n_in,
                              void* d_out, int out_size, void* d_ws, size_t ws_size,
                              hipStream_t stream) {
  (void)in_sizes; (void)n_in; (void)out_size; (void)ws_size;
  const float* x      = (const float*)d_in[0];  // (32,577,768)
  const float* w_qkv  = (const float*)d_in[1];  // (2304,768)
  const float* w_proj = (const float*)d_in[2];  // (768,768)
  float* out = (float*)d_out;                   // (32,577,768)

  constexpr size_t PL = (size_t)32 * 12 * 577 * 64;  // halves per qkv plane
  _Float16* qkv_raw  = (_Float16*)d_ws;              // 3*PL f16 (pre-RoPE)
  _Float16* qkv_rot  = qkv_raw + 3 * PL;             // 3*PL f16 (post-RoPE)
  float*    attn_out = (float*)(qkv_rot + 3 * PL);   // 32*577*768 f32

  constexpr int M = 32 * 577;  // 18464
  const dim3 gemm_block(256);

  gemm_wmma_kernel<1><<<dim3(2304 / 128, (M + 127) / 128), gemm_block, 0, stream>>>(
      x, w_qkv, qkv_raw, M, 2304, 768);

  {
    const unsigned total = 3u * (unsigned)PL;
    rope_kernel<<<(total + 255u) / 256u, 256, 0, stream>>>(qkv_raw, qkv_rot);
  }

  attn_kernel<<<dim3(10, 12, 32), dim3(128), 0, stream>>>(
      qkv_rot, qkv_rot + PL, qkv_rot + 2 * PL, attn_out);

  gemm_wmma_kernel<0><<<dim3(768 / 128, (M + 127) / 128), gemm_block, 0, stream>>>(
      attn_out, w_proj, out, M, 768, 768);
}
